// GraphStateEncoder_20572893348183
// MI455X (gfx1250) — matrix operand order repaired and checked by reading, not compiled, not run on hardware
//
#include <hip/hip_runtime.h>
#include <hip/hip_bf16.h>

// ---------------------------------------------------------------------------
// CDNA5 (gfx1250) wave32 WMMA implementation of the GNN message-passing net.
//   - all big GEMMs use v_wmma_f32_16x16x32_bf16 (fp32 -> bf16 inputs, f32 acc)
//   - msg MLP restructured:  h1 = relu(P[row] + Q[col] + ef@M_l + bvec_l)
//       P = x@W1a, Q = x@W1b  (per-node, L2 resident)
//       M_l = edge_w@W1c_l  (folded edge encoder, [32,128])
//       bvec_l = msg_b1_l + edge_b@W1c_l
//   - scatter-mean via global_atomic_add_f32, deg folded into update kernel
//   - B-fragments batch-loaded per k-step so 8 WMMAs issue back-to-back
// ---------------------------------------------------------------------------

typedef __bf16 bhalf;
typedef __attribute__((ext_vector_type(16))) __bf16 v16bf;
typedef __attribute__((ext_vector_type(8)))  float  v8f;

#define HDIM 128

__device__ __forceinline__ v8f wmma_bf16(v16bf a, v16bf b, v8f c) {
  return __builtin_amdgcn_wmma_f32_16x16x32_bf16(
      /*neg_a=*/false, a, /*neg_b=*/false, b,
      /*c_mod=*/(short)0, c, /*reuse_a=*/false, /*reuse_b=*/false);
}

// A-fragment 16x32 bf16 from fp32 row-major global memory (rows clamped).
// ISA layout: lane m=lane&15, g=lane>>4; halves 0..7 -> k=8g+h, 8..15 -> k=16+8g+(h-8)
__device__ __forceinline__ v16bf load_a_frag_f32(
    const float* __restrict__ A, int lda, int row0, int rows, int k0, int lane,
    const float* __restrict__ rowscale) {
  int m = lane & 15, g = lane >> 4;
  int r = row0 + m;
  if (r >= rows) r = rows - 1;           // clamped; garbage rows masked at store
  float s = rowscale ? rowscale[r] : 1.0f;
  const float* p = A + (size_t)r * lda + k0 + 8 * g;
  v16bf a;
#pragma unroll
  for (int i = 0; i < 8; ++i) a[i] = (__bf16)(p[i] * s);
#pragma unroll
  for (int i = 0; i < 8; ++i) a[8 + i] = (__bf16)(p[16 + i] * s);
  return a;
}

// A-fragment 16x32 bf16 from an LDS tile [16][ldt] (bf16, row-major)
__device__ __forceinline__ v16bf load_a_frag_lds(
    const bhalf* T, int ldt, int k0, int lane) {
  int m = lane & 15, g = lane >> 4;
  const bhalf* p = T + m * ldt + k0 + 8 * g;
  v16bf a;
#pragma unroll
  for (int i = 0; i < 8; ++i) a[i] = p[i];
#pragma unroll
  for (int i = 0; i < 8; ++i) a[8 + i] = p[16 + i];
  return a;
}

// B-fragment 32x16 bf16 from packed transposed weights Wt[n][k] (global bf16).
// Lane n=lane&15, g=lane>>4; halves 0..7 -> k=8g+h, 8..15 -> k=16+8g+(h-8).
__device__ __forceinline__ v16bf load_b_frag_g(
    const bhalf* __restrict__ Wt, int K, int n0, int k0, int lane) {
  int n = n0 + (lane & 15), g = lane >> 4;
  const bhalf* p = Wt + (size_t)n * K + k0 + 8 * g;
  v16bf b;
#pragma unroll
  for (int i = 0; i < 8; ++i) b[i] = p[i];
#pragma unroll
  for (int i = 0; i < 8; ++i) b[8 + i] = p[16 + i];
  return b;
}

// ---------------------------------------------------------------------------
// Weight pre-pack: fp32 -> bf16, transposed [n][k]; plus fused M_l and bvec_l.
// ---------------------------------------------------------------------------
__global__ void __launch_bounds__(256) pack_weights_kernel(
    const float* __restrict__ node_w, const float* __restrict__ edge_w,
    const float* __restrict__ edge_b, const float* __restrict__ msg_w1,
    const float* __restrict__ msg_b1, const float* __restrict__ msg_w2,
    const float* __restrict__ upd_w1, const float* __restrict__ upd_w2,
    const float* __restrict__ ro_w1, const float* __restrict__ ro_w2,
    bhalf* nWt, bhalf* W1at, bhalf* W1bt, bhalf* W2t,
    bhalf* Mt, float* bvec, bhalf* U1at, bhalf* U1bt, bhalf* U2t,
    bhalf* Ro1t, bhalf* Ro2t) {
  int t = threadIdx.x;
  for (int i = t; i < 128 * 64; i += 256) {          // node_w [64,128] -> [128][64]
    int nn = i / 64, k = i % 64;
    nWt[i] = (bhalf)node_w[k * 128 + nn];
  }
  for (int l = 0; l < 3; ++l) {
    const float* w1 = msg_w1 + (size_t)l * 384 * 128;
    const float* w1c = w1 + 256 * 128;
    for (int i = t; i < 128 * 128; i += 256) {
      int nn = i >> 7, k = i & 127;
      W1at[l * 16384 + i] = (bhalf)w1[k * 128 + nn];
      W1bt[l * 16384 + i] = (bhalf)w1[(128 + k) * 128 + nn];
      W2t[l * 16384 + i]  = (bhalf)msg_w2[(size_t)l * 16384 + k * 128 + nn];
      U1at[l * 16384 + i] = (bhalf)upd_w1[(size_t)l * 256 * 128 + k * 128 + nn];
      U1bt[l * 16384 + i] = (bhalf)upd_w1[(size_t)l * 256 * 128 + (128 + k) * 128 + nn];
      U2t[l * 16384 + i]  = (bhalf)upd_w2[(size_t)l * 16384 + k * 128 + nn];
    }
    // M_l[a][n] = sum_h edge_w[a][h] * W1c[h][n]   -> Mt[n][a]
    for (int i = t; i < 128 * 32; i += 256) {
      int nn = i / 32, a = i % 32;
      float s = 0.0f;
      for (int h = 0; h < 128; ++h) s += edge_w[a * 128 + h] * w1c[h * 128 + nn];
      Mt[l * 4096 + i] = (bhalf)s;
    }
    // bvec_l[n] = msg_b1[l][n] + sum_h edge_b[h] * W1c[h][n]
    for (int nn = t; nn < 128; nn += 256) {
      float s = msg_b1[l * 128 + nn];
      for (int h = 0; h < 128; ++h) s += edge_b[h] * w1c[h * 128 + nn];
      bvec[l * 128 + nn] = s;
    }
  }
  for (int i = t; i < 128 * 128; i += 256) {
    int nn = i >> 7, k = i & 127;
    Ro1t[i] = (bhalf)ro_w1[k * 128 + nn];
    Ro2t[i] = (bhalf)ro_w2[k * 128 + nn];
  }
}

// ---------------------------------------------------------------------------
// Generic: C[M,128] = A[M,K] @ W + bias, W pre-packed bf16 [128][K].
// One wave per 16-row tile, 8 waves per block; K is compile-time so the k-loop
// unrolls and B-fragment loads pipeline under the WMMAs.
// ---------------------------------------------------------------------------
template <int K>
__global__ void __launch_bounds__(256) gemm_rows_kernel(
    const float* __restrict__ A, int lda,
    const bhalf* __restrict__ Wt, const float* __restrict__ bias,
    float* __restrict__ C, int M) {
  int lane = threadIdx.x & 31;
  int wv = threadIdx.x >> 5;
  int row0 = (blockIdx.x * 8 + wv) * 16;
  if (row0 >= M) return;
  int g = lane >> 4, n = lane & 15;
  v8f acc[8];
#pragma unroll
  for (int nt = 0; nt < 8; ++nt) {
    float bv = bias ? bias[nt * 16 + n] : 0.0f;
#pragma unroll
    for (int r = 0; r < 8; ++r) acc[nt][r] = bv;
  }
#pragma unroll
  for (int k0 = 0; k0 < K; k0 += 32) {
    v16bf a = load_a_frag_f32(A, lda, row0, M, k0, lane, nullptr);
    v16bf bfr[8];
#pragma unroll
    for (int nt = 0; nt < 8; ++nt) bfr[nt] = load_b_frag_g(Wt, K, nt * 16, k0, lane);
#pragma unroll
    for (int nt = 0; nt < 8; ++nt) acc[nt] = wmma_bf16(a, bfr[nt], acc[nt]);
  }
#pragma unroll
  for (int nt = 0; nt < 8; ++nt)
#pragma unroll
    for (int r = 0; r < 8; ++r) {
      int m = row0 + r + 8 * g;
      if (m < M) C[(size_t)m * HDIM + nt * 16 + n] = acc[nt][r];
    }
}

// ---------------------------------------------------------------------------
// Edge kernel: h1 = relu(P[row] + Q[col] + ef@M_l + bvec); msgs = h1@W2 + b2;
// atomic scatter-add into agg[col]. One wave per 16-edge tile, 4 waves/block.
// ---------------------------------------------------------------------------
__global__ void __launch_bounds__(128) edge_msg_kernel(
    const int* __restrict__ erow, const int* __restrict__ ecol,
    const float* __restrict__ ef,
    const float* __restrict__ P, const float* __restrict__ Q,
    const float* __restrict__ bvec, const float* __restrict__ b2,
    const bhalf* __restrict__ Mt, const bhalf* __restrict__ W2t,
    float* __restrict__ agg, int E) {
  __shared__ bhalf sH[4][16 * HDIM];
  int lane = threadIdx.x & 31, wv = threadIdx.x >> 5;
  int e0 = (blockIdx.x * 4 + wv) * 16;
  bool active = (e0 < E);
  int g = lane >> 4, n = lane & 15;
  int rm[8], cm[8];
  v8f acc[8];
  if (active) {
    // issue ef A-frag + all 8 M_l B-frags early; they complete under the gathers
    v16bf a = load_a_frag_f32(ef, 32, e0, E, 0, lane, nullptr);
    v16bf bfr[8];
#pragma unroll
    for (int nt = 0; nt < 8; ++nt) bfr[nt] = load_b_frag_g(Mt, 32, nt * 16, 0, lane);

    int ei = e0 + (lane & 15);
    if (ei >= E) ei = E - 1;
    int rowv = erow[ei];
    int colv = ecol[ei];
#pragma unroll
    for (int r = 0; r < 8; ++r) {                // broadcast tile indices
      rm[r] = __shfl(rowv, r + 8 * g, 32);       // -> ds_bpermute
      cm[r] = __shfl(colv, r + 8 * g, 32);
    }
    // accumulator init: gathered P[row] + Q[col] + fused bias (L2-resident)
#pragma unroll
    for (int nt = 0; nt < 8; ++nt) {
      int nn = nt * 16 + n;
      float bv = bvec[nn];
#pragma unroll
      for (int r = 0; r < 8; ++r)
        acc[nt][r] = P[(size_t)rm[r] * HDIM + nn] + Q[(size_t)cm[r] * HDIM + nn] + bv;
    }
    // + ef @ M_l   (single K=32 WMMA step)
#pragma unroll
    for (int nt = 0; nt < 8; ++nt) acc[nt] = wmma_bf16(a, bfr[nt], acc[nt]);
    // relu -> LDS (bf16) to re-fragment for second GEMM
#pragma unroll
    for (int nt = 0; nt < 8; ++nt)
#pragma unroll
      for (int r = 0; r < 8; ++r) {
        int m = r + 8 * g;
        sH[wv][m * HDIM + nt * 16 + n] = (bhalf)fmaxf(acc[nt][r], 0.0f);
      }
  }
  __syncthreads();
  if (active) {
#pragma unroll
    for (int nt = 0; nt < 8; ++nt) {
      float bv = b2[nt * 16 + n];
#pragma unroll
      for (int r = 0; r < 8; ++r) acc[nt][r] = bv;
    }
#pragma unroll
    for (int k0 = 0; k0 < HDIM; k0 += 32) {
      v16bf a = load_a_frag_lds(&sH[wv][0], HDIM, k0, lane);
      v16bf bfr[8];
#pragma unroll
      for (int nt = 0; nt < 8; ++nt) bfr[nt] = load_b_frag_g(W2t, HDIM, nt * 16, k0, lane);
#pragma unroll
      for (int nt = 0; nt < 8; ++nt) acc[nt] = wmma_bf16(a, bfr[nt], acc[nt]);
    }
    // scatter-add messages into agg[col]
#pragma unroll
    for (int nt = 0; nt < 8; ++nt)
#pragma unroll
      for (int r = 0; r < 8; ++r) {
        int m = r + 8 * g;
        if (e0 + m < E)
          atomicAdd(&agg[(size_t)cm[r] * HDIM + nt * 16 + n], acc[nt][r]);
      }
  }
}

// ---------------------------------------------------------------------------
// Node update: x = relu(relu(x@U1a + (agg/deg)@U1b + ub1) @ U2 + ub2), in place.
// ---------------------------------------------------------------------------
__global__ void __launch_bounds__(128) node_update_kernel(
    float* __restrict__ x, const float* __restrict__ agg,
    const float* __restrict__ invdeg,
    const bhalf* __restrict__ U1at, const bhalf* __restrict__ U1bt,
    const bhalf* __restrict__ U2t,
    const float* __restrict__ ub1, const float* __restrict__ ub2, int Nn) {
  __shared__ bhalf sH[4][16 * HDIM];
  int lane = threadIdx.x & 31, wv = threadIdx.x >> 5;
  int row0 = (blockIdx.x * 4 + wv) * 16;
  bool active = (row0 < Nn);
  int g = lane >> 4, n = lane & 15;
  v8f acc[8];
  if (active) {
#pragma unroll
    for (int nt = 0; nt < 8; ++nt) {
      float bv = ub1[nt * 16 + n];
#pragma unroll
      for (int r = 0; r < 8; ++r) acc[nt][r] = bv;
    }
#pragma unroll
    for (int k0 = 0; k0 < HDIM; k0 += 32) {
      {
        v16bf ax = load_a_frag_f32(x, HDIM, row0, Nn, k0, lane, nullptr);
        v16bf bfr[8];
#pragma unroll
        for (int nt = 0; nt < 8; ++nt) bfr[nt] = load_b_frag_g(U1at, HDIM, nt * 16, k0, lane);
#pragma unroll
        for (int nt = 0; nt < 8; ++nt) acc[nt] = wmma_bf16(ax, bfr[nt], acc[nt]);
      }
      {
        v16bf aa = load_a_frag_f32(agg, HDIM, row0, Nn, k0, lane, invdeg);  // deg fold
        v16bf bfr[8];
#pragma unroll
        for (int nt = 0; nt < 8; ++nt) bfr[nt] = load_b_frag_g(U1bt, HDIM, nt * 16, k0, lane);
#pragma unroll
        for (int nt = 0; nt < 8; ++nt) acc[nt] = wmma_bf16(aa, bfr[nt], acc[nt]);
      }
    }
#pragma unroll
    for (int nt = 0; nt < 8; ++nt)
#pragma unroll
      for (int r = 0; r < 8; ++r) {
        int m = r + 8 * g;
        sH[wv][m * HDIM + nt * 16 + n] = (bhalf)fmaxf(acc[nt][r], 0.0f);
      }
  }
  __syncthreads();
  if (active) {
#pragma unroll
    for (int nt = 0; nt < 8; ++nt) {
      float bv = ub2[nt * 16 + n];
#pragma unroll
      for (int r = 0; r < 8; ++r) acc[nt][r] = bv;
    }
#pragma unroll
    for (int k0 = 0; k0 < HDIM; k0 += 32) {
      v16bf a = load_a_frag_lds(&sH[wv][0], HDIM, k0, lane);
      v16bf bfr[8];
#pragma unroll
      for (int nt = 0; nt < 8; ++nt) bfr[nt] = load_b_frag_g(U2t, HDIM, nt * 16, k0, lane);
#pragma unroll
      for (int nt = 0; nt < 8; ++nt) acc[nt] = wmma_bf16(a, bfr[nt], acc[nt]);
    }
#pragma unroll
    for (int nt = 0; nt < 8; ++nt)
#pragma unroll
      for (int r = 0; r < 8; ++r) {
        int m = row0 + r + 8 * g;
        if (m < Nn)
          x[(size_t)m * HDIM + nt * 16 + n] = fmaxf(acc[nt][r], 0.0f);
      }
  }
}

// ---------------------------------------------------------------------------
// Degree / pooling helpers.
// ---------------------------------------------------------------------------
__global__ void deg_accum_kernel(const int* __restrict__ ecol,
                                 float* __restrict__ deg, int E) {
  int i = blockIdx.x * blockDim.x + threadIdx.x;
  if (i < E) atomicAdd(&deg[ecol[i]], 1.0f);
}

__global__ void invdeg_kernel(float* __restrict__ deg, int Nn) {
  int i = blockIdx.x * blockDim.x + threadIdx.x;
  if (i < Nn) deg[i] = 1.0f / fmaxf(deg[i], 1.0f);
}

__global__ void pool_accum_kernel(const float* __restrict__ x,
                                  const int* __restrict__ batch,
                                  float* __restrict__ pooled,
                                  float* __restrict__ gcnt, int Nn) {
  int i = blockIdx.x * blockDim.x + threadIdx.x;
  if (i < Nn * HDIM) {
    int node = i >> 7, c = i & 127;
    int b = batch[node];
    atomicAdd(&pooled[b * HDIM + c], x[i]);
    if (c == 0) atomicAdd(&gcnt[b], 1.0f);
  }
}

__global__ void pool_finalize_kernel(float* __restrict__ pooled,
                                     const float* __restrict__ gcnt) {
  int i = blockIdx.x * blockDim.x + threadIdx.x;
  if (i < 16 * HDIM) pooled[i] /= fmaxf(gcnt[i >> 7], 1.0f);
}

// ---------------------------------------------------------------------------
// Readout: out[16,128] = relu(pooled@Ro1 + rb1) @ Ro2 + rb2 — one wave, G==16.
// ---------------------------------------------------------------------------
__global__ void __launch_bounds__(32) readout_kernel(
    const float* __restrict__ pooled,
    const bhalf* __restrict__ Ro1t, const bhalf* __restrict__ Ro2t,
    const float* __restrict__ rb1, const float* __restrict__ rb2,
    float* __restrict__ out) {
  __shared__ bhalf sH[16 * HDIM];
  int lane = threadIdx.x;
  int g = lane >> 4, n = lane & 15;
  v8f acc[8];
#pragma unroll
  for (int nt = 0; nt < 8; ++nt) {
    float bv = rb1[nt * 16 + n];
#pragma unroll
    for (int r = 0; r < 8; ++r) acc[nt][r] = bv;
  }
#pragma unroll
  for (int k0 = 0; k0 < HDIM; k0 += 32) {
    v16bf a = load_a_frag_f32(pooled, HDIM, 0, 16, k0, lane, nullptr);
    v16bf bfr[8];
#pragma unroll
    for (int nt = 0; nt < 8; ++nt) bfr[nt] = load_b_frag_g(Ro1t, HDIM, nt * 16, k0, lane);
#pragma unroll
    for (int nt = 0; nt < 8; ++nt) acc[nt] = wmma_bf16(a, bfr[nt], acc[nt]);
  }
#pragma unroll
  for (int nt = 0; nt < 8; ++nt)
#pragma unroll
    for (int r = 0; r < 8; ++r)
      sH[(r + 8 * g) * HDIM + nt * 16 + n] = (bhalf)fmaxf(acc[nt][r], 0.0f);
  __syncthreads();
#pragma unroll
  for (int nt = 0; nt < 8; ++nt) {
    float bv = rb2[nt * 16 + n];
#pragma unroll
    for (int r = 0; r < 8; ++r) acc[nt][r] = bv;
  }
#pragma unroll
  for (int k0 = 0; k0 < HDIM; k0 += 32) {
    v16bf a = load_a_frag_lds(sH, HDIM, k0, lane);
    v16bf bfr[8];
#pragma unroll
    for (int nt = 0; nt < 8; ++nt) bfr[nt] = load_b_frag_g(Ro2t, HDIM, nt * 16, k0, lane);
#pragma unroll
    for (int nt = 0; nt < 8; ++nt) acc[nt] = wmma_bf16(a, bfr[nt], acc[nt]);
  }
#pragma unroll
  for (int nt = 0; nt < 8; ++nt)
#pragma unroll
    for (int r = 0; r < 8; ++r)
      out[(size_t)(r + 8 * g) * HDIM + nt * 16 + n] = acc[nt][r];
}

// ---------------------------------------------------------------------------
extern "C" void kernel_launch(void* const* d_in, const int* in_sizes, int n_in,
                              void* d_out, int out_size, void* d_ws, size_t ws_size,
                              hipStream_t stream) {
  const float* node_features = (const float*)d_in[0];
  const int*   edge_index    = (const int*)d_in[1];
  const float* edge_features = (const float*)d_in[2];
  const int*   batch         = (const int*)d_in[3];
  const float* node_w = (const float*)d_in[4];
  const float* node_b = (const float*)d_in[5];
  const float* edge_w = (const float*)d_in[6];
  const float* edge_b = (const float*)d_in[7];
  const float* msg_w1 = (const float*)d_in[8];
  const float* msg_b1 = (const float*)d_in[9];
  const float* msg_w2 = (const float*)d_in[10];
  const float* msg_b2 = (const float*)d_in[11];
  const float* upd_w1 = (const float*)d_in[12];
  const float* upd_b1 = (const float*)d_in[13];
  const float* upd_w2 = (const float*)d_in[14];
  const float* upd_b2 = (const float*)d_in[15];
  const float* ro_w1  = (const float*)d_in[16];
  const float* ro_b1  = (const float*)d_in[17];
  const float* ro_w2  = (const float*)d_in[18];
  const float* ro_b2  = (const float*)d_in[19];

  const int Nn = in_sizes[3];          // batch has N elements
  const int E  = in_sizes[1] / 2;      // edge_index is [2, E]
  const int* erow = edge_index;
  const int* ecol = edge_index + E;

  // ---- workspace carve-up (256B aligned regions) ----
  char* wsp = (char*)d_ws;
  auto carve = [&](size_t bytes) -> void* {
    void* p = (void*)wsp;
    wsp += (bytes + 255) & ~(size_t)255;
    return p;
  };
  float* x      = (float*)carve((size_t)Nn * HDIM * 4);
  float* P      = (float*)carve((size_t)Nn * HDIM * 4);
  float* Q      = (float*)carve((size_t)Nn * HDIM * 4);
  float* agg    = (float*)carve((size_t)Nn * HDIM * 4);
  float* invdeg = (float*)carve((size_t)Nn * 4);
  float* pooled = (float*)carve(16 * HDIM * 4);
  float* gcnt   = (float*)carve(16 * 4);
  float* bvec   = (float*)carve(3 * HDIM * 4);
  bhalf* nWt    = (bhalf*)carve(128 * 64 * 2);
  bhalf* W1at   = (bhalf*)carve(3 * 16384 * 2);
  bhalf* W1bt   = (bhalf*)carve(3 * 16384 * 2);
  bhalf* W2t    = (bhalf*)carve(3 * 16384 * 2);
  bhalf* Mt     = (bhalf*)carve(3 * 4096 * 2);
  bhalf* U1at   = (bhalf*)carve(3 * 16384 * 2);
  bhalf* U1bt   = (bhalf*)carve(3 * 16384 * 2);
  bhalf* U2t    = (bhalf*)carve(3 * 16384 * 2);
  bhalf* Ro1t   = (bhalf*)carve(16384 * 2);
  bhalf* Ro2t   = (bhalf*)carve(16384 * 2);

  // ---- one-time prep (inside capture; deterministic) ----
  hipMemsetAsync(invdeg, 0, (size_t)Nn * 4, stream);
  hipMemsetAsync(pooled, 0, 16 * HDIM * 4, stream);
  hipMemsetAsync(gcnt, 0, 16 * 4, stream);

  pack_weights_kernel<<<1, 256, 0, stream>>>(
      node_w, edge_w, edge_b, msg_w1, msg_b1, msg_w2, upd_w1, upd_w2,
      ro_w1, ro_w2, nWt, W1at, W1bt, W2t, Mt, bvec, U1at, U1bt, U2t, Ro1t, Ro2t);

  deg_accum_kernel<<<(E + 255) / 256, 256, 0, stream>>>(ecol, invdeg, E);
  invdeg_kernel<<<(Nn + 255) / 256, 256, 0, stream>>>(invdeg, Nn);

  // x = node_features @ node_w + node_b   (K = 64)
  int ntiles = (Nn + 15) / 16;
  gemm_rows_kernel<64><<<(ntiles + 7) / 8, 256, 0, stream>>>(
      node_features, 64, nWt, node_b, x, Nn);

  int etiles = (E + 15) / 16;
  for (int l = 0; l < 3; ++l) {
    // P = x @ W1a, Q = x @ W1b  (no bias)
    gemm_rows_kernel<128><<<(ntiles + 7) / 8, 256, 0, stream>>>(
        x, HDIM, W1at + (size_t)l * 16384, nullptr, P, Nn);
    gemm_rows_kernel<128><<<(ntiles + 7) / 8, 256, 0, stream>>>(
        x, HDIM, W1bt + (size_t)l * 16384, nullptr, Q, Nn);
    hipMemsetAsync(agg, 0, (size_t)Nn * HDIM * 4, stream);
    edge_msg_kernel<<<(etiles + 3) / 4, 128, 0, stream>>>(
        erow, ecol, edge_features, P, Q, bvec + l * HDIM, msg_b2 + l * HDIM,
        Mt + (size_t)l * 4096, W2t + (size_t)l * 16384, agg, E);
    node_update_kernel<<<(ntiles + 3) / 4, 128, 0, stream>>>(
        x, agg, invdeg, U1at + (size_t)l * 16384, U1bt + (size_t)l * 16384,
        U2t + (size_t)l * 16384, upd_b1 + l * HDIM, upd_b2 + l * HDIM, Nn);
  }

  pool_accum_kernel<<<((Nn * HDIM) + 255) / 256, 256, 0, stream>>>(
      x, batch, pooled, gcnt, Nn);
  pool_finalize_kernel<<<8, 256, 0, stream>>>(pooled, gcnt);

  readout_kernel<<<1, 32, 0, stream>>>(pooled, Ro1t, Ro2t, ro_b1, ro_b2,
                                       (float*)d_out);
}
